// LocAtt2_20229295964634
// MI455X (gfx1250) — compile-verified, matrix operand order checked by reading, not run on hardware
//
#include <hip/hip_runtime.h>

// ---- problem constants (fixed by setup_inputs) ----
#define B_IMG 8
#define C_CH  64
#define H_IMG 128
#define W_IMG 128
#define TILE  16
#define HALO  18                 // TILE + 2 (3x3 neighborhood, pad=1)
#define NPIX  (HALO*HALO)        // 324 valid halo pixels
#define PPIX  336                // padded to multiple of 16 for M-tiling
#define XSTR  72                 // bf16 channel stride per pixel (144B, 16B aligned)
#define WSTR  72                 // bf16 k-stride per output channel row
#define OSTR  68                 // f32 stride per pixel in q/k/v (16B-aligned float4;
                                 // D-store halves land on disjoint banks: 8*68%64=32)

// ---- LDS layout (bytes) ----
#define X_OFF   0
#define X_BYTES (PPIX*XSTR*2)        // 48384
#define W_OFF   (X_OFF + X_BYTES)
#define W_BYTES (3*C_CH*WSTR*2)      // 27648
#define Q_OFF   (W_OFF + W_BYTES)    // 76032
#define Q_BYTES (256*OSTR*4)         // 69632
#define K_OFF   (Q_OFF + Q_BYTES)    // 145664
#define KV_BYTES (NPIX*OSTR*4)       // 88128 (324 pixels only; tail tile predicated)
#define V_OFF   (K_OFF + KV_BYTES)   // 233792
#define LDS_TOTAL (V_OFF + KV_BYTES) // 321920 < 327680 (320KB)

typedef __attribute__((ext_vector_type(16))) __bf16 v16bf;
typedef __attribute__((ext_vector_type(8)))  float  v8f;

union V16U { uint4 u[2]; v16bf v; };

extern "C" __global__ __launch_bounds__(256, 1)
void locatt_fused(const float* __restrict__ x,
                  const float* __restrict__ wq, const float* __restrict__ bq,
                  const float* __restrict__ wk, const float* __restrict__ bk,
                  const float* __restrict__ wv, const float* __restrict__ bv,
                  float* __restrict__ out)
{
    extern __shared__ char lds[];
    __bf16* x_bf = (__bf16*)(lds + X_OFF);   // [pixel][ch] bf16, stride XSTR
    __bf16* w_bf = (__bf16*)(lds + W_OFF);   // [mat][n][k] bf16, stride WSTR
    float*  q_s  = (float*)(lds + Q_OFF);    // [interior pix][ch], stride OSTR
    float*  k_s  = (float*)(lds + K_OFF);    // [halo pix][ch]
    float*  v_s  = (float*)(lds + V_OFF);    // [halo pix][ch]

    const int tid = threadIdx.x;
    const int bx = blockIdx.x, by = blockIdx.y, bz = blockIdx.z;
    const int gx0 = bx*TILE - 1, gy0 = by*TILE - 1;

    // ---------- Phase 1a: x halo -> LDS bf16, transposed to [pix][ch] ----------
    for (int idx = tid; idx < C_CH*PPIX; idx += 256) {
        int c = idx / PPIX;
        int p = idx - c*PPIX;
        float val = 0.f;
        if (p < NPIX) {
            int hy = p / HALO, hx = p - hy*HALO;
            int gy = gy0 + hy, gx = gx0 + hx;
            if (gy >= 0 && gy < H_IMG && gx >= 0 && gx < W_IMG)
                val = x[(((size_t)bz*C_CH + c)*H_IMG + gy)*W_IMG + gx];
        }
        x_bf[p*XSTR + c] = (__bf16)val;
    }
    // ---------- Phase 1b: weights -> LDS bf16 ----------
    for (int idx = tid; idx < 3*C_CH*C_CH; idx += 256) {
        int mat = idx >> 12;           // / 4096
        int r   = idx & 4095;
        const float* w = (mat == 0) ? wq : (mat == 1 ? wk : wv);
        int n = r >> 6, k = r & 63;
        w_bf[mat*(C_CH*WSTR) + n*WSTR + k] = (__bf16)w[r];
    }
    __syncthreads();

    // ---------- Phase 2: q/k/v projections via v_wmma_f32_16x16x32_bf16 ----------
    // jobs: q -> 16 M-tiles x 4 N-tiles (interior rows), k/v -> 21 x 4 over halo.
    // 232 jobs == 8 waves * 29 (perfect balance). Job decode forced to SALU.
    const int lane = tid & 31;
    const int wave = __builtin_amdgcn_readfirstlane(tid >> 5);  // scalar wave id
    const int ln15 = lane & 15;
    const int hi   = lane >> 4;        // A/B half: lanes 16-31 hold K+8
    const int kb   = hi * 8;

    for (int j = wave; j < 232; j += 8) {
        int mat, mt, nt;
        if (j < 64) { mat = 0; mt = j >> 2; nt = j & 3; }
        else { int r = j - 64; mat = 1 + r/84; r %= 84; mt = r >> 2; nt = r & 3; }

        const int n = nt*16 + ln15;
        const float* bp = (mat == 0) ? bq : (mat == 1 ? bk : bv);
        const float bias = bp[n];
        v8f acc = {bias,bias,bias,bias,bias,bias,bias,bias};

        // A-row: q jobs address interior row mt inside the halo; k/v use flat halo idx
        const int pixA = (mat == 0) ? ((mt + 1)*HALO + 1 + ln15) : (mt*16 + ln15);
        const __bf16* arow = x_bf + pixA*XSTR;
        const __bf16* brow = w_bf + mat*(C_CH*WSTR) + n*WSTR;

        #pragma unroll
        for (int ks = 0; ks < 2; ks++) {        // K = 64 in two 32-wide steps
            const int c0 = ks*32 + kb;
            V16U A, Bf;
            A.u[0]  = *(const uint4*)(arow + c0);        // K = c0 .. c0+7
            A.u[1]  = *(const uint4*)(arow + c0 + 16);   // K = c0+16 .. c0+23
            Bf.u[0] = *(const uint4*)(brow + c0);
            Bf.u[1] = *(const uint4*)(brow + c0 + 16);
            acc = __builtin_amdgcn_wmma_f32_16x16x32_bf16(
                      false, A.v, false, Bf.v, (short)0, acc, false, false);
        }

        float* ob = (mat == 0) ? q_s : (mat == 1 ? k_s : v_s);
        const int mb = mt*16 + hi*8;           // D: VGPR r -> M = r (+8 upper half)
        const bool tail = (mat != 0) && (mt == 20);   // scalar: last halo tile
        if (!tail) {
            #pragma unroll
            for (int r8 = 0; r8 < 8; r8++)
                ob[(mb + r8)*OSTR + n] = acc[r8];
        } else {
            #pragma unroll
            for (int r8 = 0; r8 < 8; r8++)
                if (mb + r8 < NPIX)                    // drop pad pixels 324..335
                    ob[(mb + r8)*OSTR + n] = acc[r8];
        }
    }
    __syncthreads();

    // ---------- Phase 3: 3x3 local attention, one thread per interior pixel ----------
    const int iy = tid >> 4, ix = tid & 15;
    int hp[9];
    #pragma unroll
    for (int p = 0; p < 9; p++) {
        int dy = p/3 - 1, dx = p - (p/3)*3 - 1;        // unfold order: p = i*3 + j
        hp[p] = (iy + 1 + dy)*HALO + (ix + 1 + dx);
    }

    float a9[9] = {0,0,0,0,0,0,0,0,0};
    const float* qrow = q_s + tid*OSTR;
    for (int c = 0; c < C_CH; c += 4) {                // float4 LDS loads (16B aligned)
        float4 q4 = *(const float4*)(qrow + c);
        #pragma unroll
        for (int p = 0; p < 9; p++) {
            float4 k4 = *(const float4*)(k_s + hp[p]*OSTR + c);
            a9[p] = fmaf(q4.x, k4.x, a9[p]);
            a9[p] = fmaf(q4.y, k4.y, a9[p]);
            a9[p] = fmaf(q4.z, k4.z, a9[p]);
            a9[p] = fmaf(q4.w, k4.w, a9[p]);
        }
    }

    float mx = a9[0];
    #pragma unroll
    for (int p = 1; p < 9; p++) mx = fmaxf(mx, a9[p]);
    float s = 0.f;
    #pragma unroll
    for (int p = 0; p < 9; p++) { a9[p] = __expf(a9[p] - mx); s += a9[p]; }
    float rs = __builtin_amdgcn_rcpf(s);
    #pragma unroll
    for (int p = 0; p < 9; p++) a9[p] *= rs;

    const int gy = by*TILE + iy, gx = bx*TILE + ix;
    float* orow = out + (((size_t)bz*C_CH)*H_IMG + gy)*W_IMG + gx;
    for (int c = 0; c < C_CH; c += 4) {
        float4 o = {0.f, 0.f, 0.f, 0.f};
        #pragma unroll
        for (int p = 0; p < 9; p++) {
            float4 v4 = *(const float4*)(v_s + hp[p]*OSTR + c);
            o.x = fmaf(a9[p], v4.x, o.x);
            o.y = fmaf(a9[p], v4.y, o.y);
            o.z = fmaf(a9[p], v4.z, o.z);
            o.w = fmaf(a9[p], v4.w, o.w);
        }
        orow[(size_t)(c + 0) * (H_IMG*W_IMG)] = o.x;   // coalesced across lanes
        orow[(size_t)(c + 1) * (H_IMG*W_IMG)] = o.y;
        orow[(size_t)(c + 2) * (H_IMG*W_IMG)] = o.z;
        orow[(size_t)(c + 3) * (H_IMG*W_IMG)] = o.w;
    }
}

extern "C" void kernel_launch(void* const* d_in, const int* in_sizes, int n_in,
                              void* d_out, int out_size, void* d_ws, size_t ws_size,
                              hipStream_t stream) {
    const float* x  = (const float*)d_in[0];
    const float* wq = (const float*)d_in[1];
    const float* bq = (const float*)d_in[2];
    const float* wk = (const float*)d_in[3];
    const float* bk = (const float*)d_in[4];
    const float* wv = (const float*)d_in[5];
    const float* bv = (const float*)d_in[6];

    dim3 grid(W_IMG/TILE, H_IMG/TILE, B_IMG);   // (8, 8, 8)
    locatt_fused<<<grid, 256, LDS_TOTAL, stream>>>(x, wq, bq, wk, bk, wv, bv,
                                                   (float*)d_out);
}
